// Net_vanilla_CNN_convert_71244917506249
// MI455X (gfx1250) — matrix operand order repaired and checked by reading, not compiled, not run on hardware
//
#include <hip/hip_runtime.h>
#include <hip/hip_bf16.h>
#include <math.h>

typedef __attribute__((ext_vector_type(16))) _Float16 v16h;
typedef __attribute__((ext_vector_type(8)))  float    v8f;

// ---------------------------------------------------------------------------
// GMU prep: per output channel o build design matrix X (25x4, ones column +
// 3 weight slices), cov = X^T X, invert 4x4 (Gauss-Jordan w/ pivoting),
// M = cov^-1 X^T (4x25). Store M and X for the forward pass.
// ---------------------------------------------------------------------------
__global__ void k_gmu_prep(const float* __restrict__ gmu_w,
                           float* __restrict__ Mmat,   // [64][4][25]
                           float* __restrict__ Xmat)   // [64][25][4]
{
    int o = threadIdx.x;
    if (o >= 64) return;
    float X[25][4];
    for (int p = 0; p < 25; ++p) {
        X[p][0] = 1.0f;
        for (int s = 0; s < 3; ++s)
            X[p][1 + s] = gmu_w[(o * 25 + p) * 3 + s];
    }
    float A[4][8];
    for (int i = 0; i < 4; ++i)
        for (int j = 0; j < 4; ++j) {
            float s = 0.f;
            for (int p = 0; p < 25; ++p) s += X[p][i] * X[p][j];
            A[i][j] = s;
        }
    for (int i = 0; i < 4; ++i)
        for (int j = 0; j < 4; ++j) A[i][4 + j] = (i == j) ? 1.f : 0.f;
    for (int col = 0; col < 4; ++col) {
        int piv = col;
        for (int r = col + 1; r < 4; ++r)
            if (fabsf(A[r][col]) > fabsf(A[piv][col])) piv = r;
        if (piv != col)
            for (int j = 0; j < 8; ++j) { float t = A[col][j]; A[col][j] = A[piv][j]; A[piv][j] = t; }
        float inv = 1.0f / A[col][col];
        for (int j = 0; j < 8; ++j) A[col][j] *= inv;
        for (int r = 0; r < 4; ++r) {
            if (r == col) continue;
            float f = A[r][col];
            for (int j = 0; j < 8; ++j) A[r][j] -= f * A[col][j];
        }
    }
    for (int c = 0; c < 4; ++c)
        for (int p = 0; p < 25; ++p) {
            float s = 0.f;
            for (int d = 0; d < 4; ++d) s += A[c][4 + d] * X[p][d];
            Mmat[o * 100 + c * 25 + p] = s;
        }
    for (int p = 0; p < 25; ++p)
        for (int c = 0; c < 4; ++c)
            Xmat[o * 100 + p * 4 + c] = X[p][c];
}

// ---------------------------------------------------------------------------
// GMU forward: one thread per (batch, pixel). 5x5 patch (pad 2), unbiased-std
// normalize, then per output channel: w = M y, err = mean((y - X w)^2),
// a = (exp(-err) - e^-1)/(1 - e^-1) - 0.5.  M/X staged in LDS (51.2 KB).
// ---------------------------------------------------------------------------
__global__ void k_gmu_forward(const float* __restrict__ x,   // (64,1,32,32)
                              const float* __restrict__ Mmat,
                              const float* __restrict__ Xmat,
                              float* __restrict__ a)          // (64,64,32,32)
{
    __shared__ float sM[6400];
    __shared__ float sX[6400];
    for (int i = threadIdx.x; i < 6400; i += blockDim.x) { sM[i] = Mmat[i]; sX[i] = Xmat[i]; }
    __syncthreads();
    int b = blockIdx.y;
    int l = blockIdx.x * blockDim.x + threadIdx.x;  // 0..1023
    int py = l >> 5, px = l & 31;
    const float* xb = x + (size_t)b * 1024;
    float yv[25];
    float sum = 0.f;
#pragma unroll
    for (int ky = 0; ky < 5; ++ky)
#pragma unroll
        for (int kx = 0; kx < 5; ++kx) {
            int yy = py + ky - 2, xx = px + kx - 2;
            float v = 0.f;
            if (yy >= 0 && yy < 32 && xx >= 0 && xx < 32) v = xb[yy * 32 + xx];
            yv[ky * 5 + kx] = v; sum += v;
        }
    float mean = sum * (1.0f / 25.0f);
    float ss = 0.f;
#pragma unroll
    for (int p = 0; p < 25; ++p) { float d = yv[p] - mean; ss += d * d; }
    float rstd = 1.0f / sqrtf(ss * (1.0f / 24.0f));   // ddof=1
#pragma unroll
    for (int p = 0; p < 25; ++p) yv[p] *= rstd;
    const float C1 = 0.36787944117144233f;            // e^-1
    const float C2 = 1.0f / (1.0f - C1);
    for (int o = 0; o < 64; ++o) {
        const float* Mo = &sM[o * 100];
        const float* Xo = &sX[o * 100];
        float w0 = 0, w1 = 0, w2 = 0, w3 = 0;
#pragma unroll
        for (int p = 0; p < 25; ++p) {
            float t = yv[p];
            w0 += Mo[p] * t; w1 += Mo[25 + p] * t; w2 += Mo[50 + p] * t; w3 += Mo[75 + p] * t;
        }
        float err = 0.f;
#pragma unroll
        for (int p = 0; p < 25; ++p) {
            float pr = Xo[p * 4] * w0 + Xo[p * 4 + 1] * w1 + Xo[p * 4 + 2] * w2 + Xo[p * 4 + 3] * w3;
            float d = yv[p] - pr; err += d * d;
        }
        err *= (1.0f / 25.0f);
        a[((size_t)(b * 64 + o)) * 1024 + l] = (expf(-err) - C1) * C2 - 0.5f;
    }
}

// ---------------------------------------------------------------------------
// 2x2 average pool, stride 2
// ---------------------------------------------------------------------------
__global__ void k_avgpool2(const float* __restrict__ in, float* __restrict__ out,
                           int BC, int H, int W)
{
    int Ho = H >> 1, Wo = W >> 1;
    size_t total = (size_t)BC * Ho * Wo;
    size_t i = blockIdx.x * (size_t)blockDim.x + threadIdx.x;
    if (i >= total) return;
    int wo = (int)(i % Wo); int ho = (int)((i / Wo) % Ho); int bc = (int)(i / ((size_t)Wo * Ho));
    const float* p = in + ((size_t)bc * H + 2 * ho) * W + 2 * wo;
    out[i] = 0.25f * (p[0] + p[1] + p[W] + p[W + 1]);
}

// ---------------------------------------------------------------------------
// BatchNorm (training stats, biased variance): per-channel scale/shift
// ---------------------------------------------------------------------------
__global__ void k_bn_stats(const float* __restrict__ in, const float* __restrict__ g,
                           const float* __restrict__ bb, float* __restrict__ scale,
                           float* __restrict__ shift, int Bn, int C, int HW)
{
    int c = blockIdx.x;
    __shared__ float ssum[256], ssq[256];
    float s = 0.f, q = 0.f;
    for (int t = threadIdx.x; t < Bn * HW; t += blockDim.x) {
        int b = t / HW, i = t % HW;
        float v = in[((size_t)b * C + c) * HW + i];
        s += v; q += v * v;
    }
    ssum[threadIdx.x] = s; ssq[threadIdx.x] = q;
    __syncthreads();
    for (int st = blockDim.x >> 1; st > 0; st >>= 1) {
        if ((int)threadIdx.x < st) { ssum[threadIdx.x] += ssum[threadIdx.x + st]; ssq[threadIdx.x] += ssq[threadIdx.x + st]; }
        __syncthreads();
    }
    if (threadIdx.x == 0) {
        float n = (float)(Bn * HW);
        float mean = ssum[0] / n;
        float var = ssq[0] / n - mean * mean;
        float sc = g[c] * rsqrtf(var + 1e-5f);
        scale[c] = sc;
        shift[c] = bb[c] - mean * sc;
    }
}

__global__ void k_bn_apply(float* __restrict__ x, const float* __restrict__ scale,
                           const float* __restrict__ shift, int C, int HW,
                           size_t total, int relu)
{
    size_t i = blockIdx.x * (size_t)blockDim.x + threadIdx.x;
    if (i >= total) return;
    int c = (int)((i / HW) % C);
    float v = x[i] * scale[c] + shift[c];
    if (relu) v = fmaxf(v, 0.f);
    x[i] = v;
}

// ===========================================================================
// WMMA fragment-packed operand preparation.
//
// Packed layout (wave32, 16x16x32 f16; contiguous k order, same for A and B):
//   operand element for (tile, kk, lane, e) lives at
//       packed[(((size_t)tile * KK + kk) * 32 + lane) * 16 + e]
//   so each lane's 16 halfs (one v16h = 8 VGPRs) are one contiguous 32 B
//   chunk -> the GEMM loads each fragment with two global_load_b128.
//
//   A fragment:  m = tile*16 + (lane&15)
//                k = kk*32 + (lane>>4)*16 + e
//   B fragment:  n = tile*16 + (lane&15)
//                k = kk*32 + (lane>>4)*16 + e
// ===========================================================================

// weights (Mv x K f32 row-major, rows >= Mv zero-padded) -> packed A
__global__ void k_pack_w(const float* __restrict__ in, _Float16* __restrict__ out,
                         int Mv, int KK, size_t total)
{
    size_t idx = blockIdx.x * (size_t)blockDim.x + threadIdx.x;
    if (idx >= total) return;
    int e = (int)(idx & 15);
    int lane = (int)((idx >> 4) & 31);
    size_t r = idx >> 9;                 // tm*KK + kk
    int kk = (int)(r % KK);
    int tm = (int)(r / KK);
    int hi = lane >> 4;
    int m = tm * 16 + (lane & 15);
    int k = kk * 32 + hi * 16 + e;
    int K = KK * 32;
    float v = (m < Mv) ? in[(size_t)m * K + k] : 0.f;
    out[idx] = (_Float16)v;
}

// implicit im2col (3x3, pad 1) of f32 activations -> packed B
// K = Cin*9, N = Bn*H*W;  k = ic*9+ky*3+kx, n = b*H*W + y*W + x
__global__ void k_im2col_pack(const float* __restrict__ in, _Float16* __restrict__ out,
                              int Bn, int Cin, int Hh, int Ww, int KK, size_t total)
{
    size_t idx = blockIdx.x * (size_t)blockDim.x + threadIdx.x;
    if (idx >= total) return;
    int e = (int)(idx & 15);
    int lane = (int)((idx >> 4) & 31);
    size_t r = idx >> 9;                 // tn*KK + kk
    int kk = (int)(r % KK);
    int tn = (int)(r / KK);
    int k = kk * 32 + (lane >> 4) * 16 + e;
    int n = tn * 16 + (lane & 15);
    int ic = k / 9, rem = k % 9;
    int ky = rem / 3, kx = rem % 3;
    int HWp = Hh * Ww;
    int b = n / HWp, hw = n % HWp;
    int y = hw / Ww, xx = hw % Ww;
    int yy = y + ky - 1, xc = xx + kx - 1;
    float v = 0.f;
    if (yy >= 0 && yy < Hh && xc >= 0 && xc < Ww)
        v = in[(((size_t)b * Cin + ic) * Hh + yy) * Ww + xc];
    out[idx] = (_Float16)v;
}

// fc path: activations (N rows x K cols, f32 row-major) -> packed B (transpose)
__global__ void k_pack_bT(const float* __restrict__ in, _Float16* __restrict__ out,
                          int KK, size_t total)
{
    size_t idx = blockIdx.x * (size_t)blockDim.x + threadIdx.x;
    if (idx >= total) return;
    int e = (int)(idx & 15);
    int lane = (int)((idx >> 4) & 31);
    size_t r = idx >> 9;
    int kk = (int)(r % KK);
    int tn = (int)(r / KK);
    int k = kk * 32 + (lane >> 4) * 16 + e;
    int n = tn * 16 + (lane & 15);
    out[idx] = (_Float16)in[(size_t)n * (KK * 32) + k];
}

// ---------------------------------------------------------------------------
// WMMA GEMM on packed fragments. One wave per (16 x 32) output strip: one
// M-tile x two N-tiles, so each A fragment feeds two v_wmma_f32_16x16x32_f16.
// K-loop unrolled 2x -> 4 static wmma; next step's operands prefetched
// (global_prefetch_b8).  C[b, m, hw] = acc + bias[m],  n = b*HW + hw.
// ---------------------------------------------------------------------------
__global__ void k_wmma_gemm_bias(const _Float16* __restrict__ Apack,
                                 const _Float16* __restrict__ Bpack,
                                 const float* __restrict__ bias,
                                 float* __restrict__ C,
                                 int KK, int ntn2, int Mv, int HW)
{
    int tile = blockIdx.x;
    int tm = tile / ntn2;
    int tn2 = 2 * (tile % ntn2);
    int lane = threadIdx.x & 31;
    int l15 = lane & 15, hi = lane >> 4;

    const v16h* Ap  = (const v16h*)Apack + ((size_t)tm * KK) * 32 + lane;
    const v16h* Bp0 = (const v16h*)Bpack + ((size_t)(tn2 + 0) * KK) * 32 + lane;
    const v16h* Bp1 = (const v16h*)Bpack + ((size_t)(tn2 + 1) * KK) * 32 + lane;

    v8f acc[2];
    acc[0] = (v8f){0.f, 0.f, 0.f, 0.f, 0.f, 0.f, 0.f, 0.f};
    acc[1] = (v8f){0.f, 0.f, 0.f, 0.f, 0.f, 0.f, 0.f, 0.f};

#pragma unroll 2
    for (int kk = 0; kk < KK; ++kk) {
        v16h af = Ap[(size_t)kk * 32];
        v16h b0 = Bp0[(size_t)kk * 32];
        v16h b1 = Bp1[(size_t)kk * 32];
        if (kk + 1 < KK) {
            __builtin_prefetch(&Ap[(size_t)(kk + 1) * 32], 0, 3);
            __builtin_prefetch(&Bp0[(size_t)(kk + 1) * 32], 0, 3);
            __builtin_prefetch(&Bp1[(size_t)(kk + 1) * 32], 0, 3);
        }
        acc[0] = __builtin_amdgcn_wmma_f32_16x16x32_f16(
            false, af, false, b0, (short)0, acc[0], false, false);
        acc[1] = __builtin_amdgcn_wmma_f32_16x16x32_f16(
            false, af, false, b1, (short)0, acc[1], false, false);
    }

#pragma unroll
    for (int t = 0; t < 2; ++t) {
        int n = (tn2 + t) * 16 + l15;
        int b = n / HW, hw = n % HW;
#pragma unroll
        for (int r = 0; r < 8; ++r) {
            int m = tm * 16 + r + 8 * hi;
            if (m < Mv)
                C[((size_t)b * Mv + m) * HW + hw] = acc[t][r] + bias[m];
        }
    }
}

// ---------------------------------------------------------------------------
// Max pool, stride == k; cells outside the image skipped (== -inf padding).
// Optionally mirrors the result into a second buffer (x_errs output).
// ---------------------------------------------------------------------------
__global__ void k_maxpool(const float* __restrict__ in, float* __restrict__ out,
                          float* __restrict__ out2,
                          int BC, int H, int W, int k, int pad, int Ho, int Wo)
{
    size_t total = (size_t)BC * Ho * Wo;
    size_t i = blockIdx.x * (size_t)blockDim.x + threadIdx.x;
    if (i >= total) return;
    int wo = (int)(i % Wo); int ho = (int)((i / Wo) % Ho); int bc = (int)(i / ((size_t)Wo * Ho));
    int y0 = ho * k - pad, x0 = wo * k - pad;
    float m = -3.402823466e38f;
    for (int dy = 0; dy < k; ++dy) {
        int y = y0 + dy; if (y < 0 || y >= H) continue;
        for (int dx = 0; dx < k; ++dx) {
            int x = x0 + dx; if (x < 0 || x >= W) continue;
            m = fmaxf(m, in[((size_t)bc * H + y) * W + x]);
        }
    }
    out[i] = m;
    if (out2) out2[i] = m;
}

// ---------------------------------------------------------------------------
// Host-side orchestration
// ---------------------------------------------------------------------------
extern "C" void kernel_launch(void* const* d_in, const int* in_sizes, int n_in,
                              void* d_out, int out_size, void* d_ws, size_t ws_size,
                              hipStream_t stream)
{
    const float* x       = (const float*)d_in[0];
    const float* gmu_w   = (const float*)d_in[1];
    const float* conv2_w = (const float*)d_in[2];
    const float* conv2_b = (const float*)d_in[3];
    const float* conv3_w = (const float*)d_in[4];
    const float* conv3_b = (const float*)d_in[5];
    const float* conv4_w = (const float*)d_in[6];
    const float* conv4_b = (const float*)d_in[7];
    const float* fc1_w   = (const float*)d_in[8];
    const float* fc1_b   = (const float*)d_in[9];
    const float* fc2_w   = (const float*)d_in[10];
    const float* fc2_b   = (const float*)d_in[11];
    const float* bn1_g   = (const float*)d_in[12]; const float* bn1_b  = (const float*)d_in[13];
    const float* bn2_g   = (const float*)d_in[14]; const float* bn2_b  = (const float*)d_in[15];
    const float* bn3_g   = (const float*)d_in[16]; const float* bn3_b  = (const float*)d_in[17];
    const float* bn4_g   = (const float*)d_in[18]; const float* bn4_b  = (const float*)d_in[19];
    const float* bnfc_g  = (const float*)d_in[20]; const float* bnfc_b = (const float*)d_in[21];

    constexpr size_t MB = 1ull << 20;
    char* ws = (char*)d_ws;
    // small persistent region (< 1 MB)
    float*    Mmat  = (float*)(ws + 0);             // 25.6 KB
    float*    Xmat  = (float*)(ws + 32 * 1024);     // 25.6 KB
    float*    scale = (float*)(ws + 64 * 1024);
    float*    shift = (float*)(ws + 66 * 1024);
    _Float16* A2    = (_Float16*)(ws + 128 * 1024); // 128x576 packed
    _Float16* A34   = (_Float16*)(ws + 320 * 1024); // 128x1152 packed (conv3 then conv4)
    _Float16* Afc1  = (_Float16*)(ws + 640 * 1024); // 128x128 packed
    _Float16* Afc2  = (_Float16*)(ws + 680 * 1024); // 16x128 packed
    _Float16* Bfc   = (_Float16*)(ws + 700 * 1024); // 128x64 packed
    // big region with lifetime-based reuse (peak ~34 MB total)
    char*  big  = ws + MB;
    float* a_buf = (float*)big;                      // GMU activations, 16.8 MB
    _Float16* Bcol = (_Float16*)big;                 // packed B buffers (reuse a_buf slot)
    float* hbuf = (float*)(big + 19 * MB);           // pooled activations h1/h2/h3
    float* cbuf = (float*)(big + 24 * MB);           // conv outputs c2/c3/c4 (<= 8.4 MB)
    float* xe   = (float*)(big + 19 * MB);           // x_errs (overwrites dead h3)
    float* f1   = (float*)(big + 19 * MB + 64 * 1024);

    float* out_xm = (float*)d_out;        // (64,10)
    float* out_xe = (float*)d_out + 640;  // (64,128,1,1)

    // ---- GMU layer
    k_gmu_prep<<<1, 64, 0, stream>>>(gmu_w, Mmat, Xmat);
    k_gmu_forward<<<dim3(4, 64), 256, 0, stream>>>(x, Mmat, Xmat, a_buf);

    // ---- avgpool 32->16, bn1
    { size_t t = 64ull * 64 * 16 * 16;
      k_avgpool2<<<(unsigned)((t + 255) / 256), 256, 0, stream>>>(a_buf, hbuf, 64 * 64, 32, 32); }
    k_bn_stats<<<64, 256, 0, stream>>>(hbuf, bn1_g, bn1_b, scale, shift, 64, 64, 256);
    k_bn_apply<<<(unsigned)((64ull * 64 * 256 + 255) / 256), 256, 0, stream>>>(
        hbuf, scale, shift, 64, 256, 64ull * 64 * 256, 0);

    // ---- conv2: M=128, K=576 (KK=18), N=16384 (WMMA)
    { size_t t = 128ull * 576;
      k_pack_w<<<(unsigned)((t + 255) / 256), 256, 0, stream>>>(conv2_w, A2, 128, 18, t); }
    { size_t t = 576ull * 16384;
      k_im2col_pack<<<(unsigned)((t + 255) / 256), 256, 0, stream>>>(hbuf, Bcol, 64, 64, 16, 16, 18, t); }
    k_wmma_gemm_bias<<<8 * 512, 32, 0, stream>>>(A2, Bcol, conv2_b, cbuf, 18, 512, 128, 256);
    k_bn_stats<<<128, 256, 0, stream>>>(cbuf, bn2_g, bn2_b, scale, shift, 64, 128, 256);
    k_bn_apply<<<(unsigned)((64ull * 128 * 256 + 255) / 256), 256, 0, stream>>>(
        cbuf, scale, shift, 128, 256, 64ull * 128 * 256, 1);

    // ---- avgpool 16->8, conv3: M=128, K=1152 (KK=36), N=4096
    { size_t t = 64ull * 128 * 64;
      k_avgpool2<<<(unsigned)((t + 255) / 256), 256, 0, stream>>>(cbuf, hbuf, 64 * 128, 16, 16); }
    { size_t t = 128ull * 1152;
      k_pack_w<<<(unsigned)((t + 255) / 256), 256, 0, stream>>>(conv3_w, A34, 128, 36, t); }
    { size_t t = 1152ull * 4096;
      k_im2col_pack<<<(unsigned)((t + 255) / 256), 256, 0, stream>>>(hbuf, Bcol, 64, 128, 8, 8, 36, t); }
    k_wmma_gemm_bias<<<8 * 128, 32, 0, stream>>>(A34, Bcol, conv3_b, cbuf, 36, 128, 128, 64);
    k_bn_stats<<<128, 256, 0, stream>>>(cbuf, bn3_g, bn3_b, scale, shift, 64, 128, 64);
    k_bn_apply<<<(unsigned)((64ull * 128 * 64 + 255) / 256), 256, 0, stream>>>(
        cbuf, scale, shift, 128, 64, 64ull * 128 * 64, 1);

    // ---- maxpool k=2 pad=1: 8->5, conv4: M=128, K=1152 (KK=36), N=1600
    { size_t t = 64ull * 128 * 25;
      k_maxpool<<<(unsigned)((t + 255) / 256), 256, 0, stream>>>(
          cbuf, hbuf, (float*)nullptr, 64 * 128, 8, 8, 2, 1, 5, 5); }
    { size_t t = 128ull * 1152;
      k_pack_w<<<(unsigned)((t + 255) / 256), 256, 0, stream>>>(conv4_w, A34, 128, 36, t); }
    { size_t t = 1152ull * 1600;
      k_im2col_pack<<<(unsigned)((t + 255) / 256), 256, 0, stream>>>(hbuf, Bcol, 64, 128, 5, 5, 36, t); }
    k_wmma_gemm_bias<<<8 * 50, 32, 0, stream>>>(A34, Bcol, conv4_b, cbuf, 36, 50, 128, 25);
    k_bn_stats<<<128, 256, 0, stream>>>(cbuf, bn4_g, bn4_b, scale, shift, 64, 128, 25);
    k_bn_apply<<<(unsigned)((64ull * 128 * 25 + 255) / 256), 256, 0, stream>>>(
        cbuf, scale, shift, 128, 25, 64ull * 128 * 25, 1);

    // ---- maxpool k=4: 5->1  (x_errs output, mirrored to d_out)
    { size_t t = 64ull * 128;
      k_maxpool<<<(unsigned)((t + 255) / 256), 256, 0, stream>>>(
          cbuf, xe, out_xe, 64 * 128, 5, 5, 4, 0, 1, 1); }

    // ---- fc1 (WMMA: M=128, K=128 (KK=4), N=64) + bnfc + relu
    { size_t t = 128ull * 128;
      k_pack_w<<<(unsigned)((t + 255) / 256), 256, 0, stream>>>(fc1_w, Afc1, 128, 4, t); }
    { size_t t = 128ull * 64;
      k_pack_bT<<<(unsigned)((t + 255) / 256), 256, 0, stream>>>(xe, Bfc, 4, t); }
    k_wmma_gemm_bias<<<8 * 2, 32, 0, stream>>>(Afc1, Bfc, fc1_b, f1, 4, 2, 128, 1);
    k_bn_stats<<<128, 256, 0, stream>>>(f1, bnfc_g, bnfc_b, scale, shift, 64, 128, 1);
    k_bn_apply<<<(unsigned)((64ull * 128 + 255) / 256), 256, 0, stream>>>(
        f1, scale, shift, 128, 1, 64ull * 128, 1);

    // ---- fc2 (WMMA: M=10 padded to 16, K=128 (KK=4), N=64) -> logits
    { size_t t = 16ull * 128;
      k_pack_w<<<(unsigned)((t + 255) / 256), 256, 0, stream>>>(fc2_w, Afc2, 10, 4, t); }
    { size_t t = 128ull * 64;
      k_pack_bT<<<(unsigned)((t + 255) / 256), 256, 0, stream>>>(f1, Bfc, 4, t); }
    k_wmma_gemm_bias<<<1 * 2, 32, 0, stream>>>(Afc2, Bfc, fc2_b, out_xm, 4, 2, 10, 1);
}
